// T10_transformer_83176336655101
// MI455X (gfx1250) — matrix operand order repaired and checked by reading, not compiled, not run on hardware
//
#include <hip/hip_runtime.h>
#include <cstddef>

#define BB 16384
#define LL 32

typedef __attribute__((ext_vector_type(2)))  float    v2f;
typedef __attribute__((ext_vector_type(4)))  float    v4f;
typedef __attribute__((ext_vector_type(8)))  float    v8f;
typedef __attribute__((ext_vector_type(4)))  unsigned v4u;
typedef __attribute__((ext_vector_type(8)))  unsigned v8u;
typedef __attribute__((ext_vector_type(16))) _Float16 v16h;

#if __has_builtin(__builtin_amdgcn_tanhf)
#define TANHF(x) __builtin_amdgcn_tanhf(x)
#else
#define TANHF(x) tanhf(x)
#endif

// sched_barrier mask: VALU|SALU|VMEM may cross; WMMA (0x8), TRANS (0x400) and DS (0x80)
// may NOT. Pinning DS keeps the LDS prefetch window to one chunk (bounded registers);
// pinning WMMA/TRANS keeps the GEMM1 tile working set at 2 chunks.
#define CHUNK_FENCE() __builtin_amdgcn_sched_barrier(0x16)

// ---------------- workspace layout (bytes, all 256B aligned) ----------------
constexpr size_t OFF_W1F  = 0;        // [16][2][32][2]  f32  (W1 A-frags, 16x16x4 f32 WMMA)
constexpr size_t OFF_B1H  = 8192;     // [16][32][4]     u32  (b1 packed f16 pairs)
constexpr size_t OFF_W2F  = 16384;    // [8][32][16]     f16  (W2 A-frags, perm + K interleave)
constexpr size_t OFF_B2F  = 24576;    // [32][8]         f32  (b2 C-init, perm)
constexpr size_t OFF_WR1F = 25600;    // [13][2][32][2]  f32
constexpr size_t OFF_BR1F = 32256;    // [13][32][8]     f32
constexpr size_t OFF_WR2F = 45568;    // [13][7][32][16] f16
constexpr size_t OFF_BR2F = 138752;   // [13][32][8]     f32
constexpr size_t OFF_WR3P = 152064;   // [13][32][8]     f32
constexpr size_t OFF_XM   = 165376;   // [B][32][4]      f32  (x.fa, x.X, m.fa, m.X)
constexpr size_t OFF_Z0   = OFF_XM + (size_t)BB * LL * 4 * 4;   // [B][8] f32 slot-ordered

// output feature permutation so G lands lane-locally for the RK state update
__device__ __forceinline__ int permf(int m) { return ((m & 4) << 1) | ((m & 8) >> 1) | (m & 3); }

// ---------------- WMMA wrappers ----------------
static __device__ __forceinline__ v8f wmma4(v2f a, v2f b, v8f c) {
  return __builtin_amdgcn_wmma_f32_16x16x4_f32(false, a, false, b, (short)0, c, false, false);
}
static __device__ __forceinline__ v8f wmma16(v16h a, v16h b, v8f c) {
  return __builtin_amdgcn_wmma_f32_16x16x32_f16(false, a, false, b, (short)0, c, false, false);
}

// cvt f32 pair -> f16 pair, add packed-f16 bias, relu: 3 VALU, no canonicalize
static __device__ __forceinline__ unsigned relu_pack_bias(float a, float b, unsigned bias) {
  auto p = __builtin_amdgcn_cvt_pkrtz(a, b);
  p = p + __builtin_bit_cast(decltype(p), bias);            // v_pk_add_f16
  asm("v_pk_max_num_f16 %0, %1, 0" : "+v"(p));              // relu
  return __builtin_bit_cast(unsigned, p);
}

static __device__ __forceinline__ float gelu_exact(float x) {
  return 0.5f * x * (1.0f + erff(x * 0.70710678118654752f));
}

// ---------------- kernel 0: pack weight fragments ----------------
__global__ __launch_bounds__(256) void k_prepack(
    const float* __restrict__ W1, const float* __restrict__ b1,
    const float* __restrict__ W2, const float* __restrict__ b2,
    const float* __restrict__ Wr1, const float* __restrict__ br1,
    const float* __restrict__ Wr2, const float* __restrict__ br2,
    const float* __restrict__ Wr3, char* __restrict__ ws) {
  int tid = blockIdx.x * blockDim.x + threadIdx.x;
  int stride = gridDim.x * blockDim.x;
  float*    W1F  = (float*)(ws + OFF_W1F);
  unsigned* B1H  = (unsigned*)(ws + OFF_B1H);
  _Float16* W2F  = (_Float16*)(ws + OFF_W2F);
  float*    B2F  = (float*)(ws + OFF_B2F);
  float*    WR1F = (float*)(ws + OFF_WR1F);
  float*    BR1F = (float*)(ws + OFF_BR1F);
  _Float16* WR2F = (_Float16*)(ws + OFF_WR2F);
  float*    BR2F = (float*)(ws + OFF_BR2F);
  float*    WR3P = (float*)(ws + OFF_WR3P);

  for (int idx = tid; idx < 16 * 2 * 32; idx += stride) {  // W1 A-frags
    int t = idx >> 6, kk = (idx >> 5) & 1, lane = idx & 31;
    int M = lane & 15, hi = lane >> 4, r = 16 * t + M;
    W1F[idx * 2 + 0] = W1[r * 8 + 4 * kk + 2 * hi + 0];
    W1F[idx * 2 + 1] = W1[r * 8 + 4 * kk + 2 * hi + 1];
  }
  for (int idx = tid; idx < 16 * 32 * 4; idx += stride) {  // b1 packed-f16 pairs
    int t = idx >> 7, lane = (idx >> 2) & 31, d = idx & 3, hi = lane >> 4;
    _Float16 e0 = (_Float16)b1[16 * t + hi * 8 + 2 * d + 0];
    _Float16 e1 = (_Float16)b1[16 * t + hi * 8 + 2 * d + 1];
    unsigned u = ((unsigned)__builtin_bit_cast(unsigned short, e1) << 16) |
                 (unsigned)__builtin_bit_cast(unsigned short, e0);
    B1H[idx] = u;
  }
  for (int idx = tid; idx < 8 * 32 * 16; idx += stride) {  // W2 A-frags (perm + K interleave)
    int c = idx >> 9, lane = (idx >> 4) & 31, j = idx & 15;
    int M = lane & 15, hi = lane >> 4;
    W2F[idx] = (_Float16)W2[permf(M) * 256 + 32 * c + hi * 8 + (j & 7) + ((j >> 3) << 4)];
  }
  for (int idx = tid; idx < 32 * 8; idx += stride) {       // b2 C-frag
    int lane = idx >> 3, v = idx & 7, hi = lane >> 4;
    B2F[idx] = b2[permf(hi * 8 + v)];
  }
  for (int idx = tid; idx < 13 * 2 * 32; idx += stride) {  // Wr1 A-frags (zero-padded)
    int t = idx >> 6, kk = (idx >> 5) & 1, lane = idx & 31;
    int M = lane & 15, hi = lane >> 4, r = 16 * t + M;
    WR1F[idx * 2 + 0] = (r < 200) ? Wr1[r * 8 + 4 * kk + 2 * hi + 0] : 0.f;
    WR1F[idx * 2 + 1] = (r < 200) ? Wr1[r * 8 + 4 * kk + 2 * hi + 1] : 0.f;
  }
  for (int idx = tid; idx < 13 * 32 * 8; idx += stride) {  // br1 C-frags
    int t = idx >> 8, lane = (idx >> 3) & 31, v = idx & 7, hi = lane >> 4;
    int r = 16 * t + hi * 8 + v;
    BR1F[idx] = (r < 200) ? br1[r] : 0.f;
  }
  for (int idx = tid; idx < 13 * 7 * 32 * 16; idx += stride) {  // Wr2 A-frags (K interleave)
    int t = idx / (7 * 512), rem = idx % (7 * 512);
    int c = rem >> 9, lane = (rem >> 4) & 31, j = rem & 15;
    int M = lane & 15, hi = lane >> 4;
    int row = 16 * t + M, col = 32 * c + hi * 8 + (j & 7) + ((j >> 3) << 4);
    WR2F[idx] = (row < 200 && col < 200) ? (_Float16)Wr2[row * 200 + col] : (_Float16)0.f;
  }
  for (int idx = tid; idx < 13 * 32 * 8; idx += stride) {  // br2 C-frags
    int t = idx >> 8, lane = (idx >> 3) & 31, v = idx & 7, hi = lane >> 4;
    int r = 16 * t + hi * 8 + v;
    BR2F[idx] = (r < 200) ? br2[r] : 0.f;
  }
  for (int idx = tid; idx < 13 * 32 * 8; idx += stride) {  // Wr3 segments
    int t = idx >> 8, lane = (idx >> 3) & 31, v = idx & 7, hi = lane >> 4;
    int r = 16 * t + hi * 8 + v;
    WR3P[idx] = (r < 200) ? Wr3[r] : 0.f;
  }
}

// ---------------- kernel 1: per-row preprocessing -> xm, z0 ----------------
__global__ __launch_bounds__(256) void k_prep(
    const float* __restrict__ X, const float* __restrict__ Fa,
    const float* __restrict__ Wi, const float* __restrict__ bi,
    const int* __restrict__ flen, char* __restrict__ ws) {
  int b = blockIdx.x * blockDim.x + threadIdx.x;
  if (b >= BB) return;
  const float* Xi = X + b * LL;
  const float* Fi = Fa + b * LL;
  int fl = flen[b];
  float flf = (float)fl;
  float s = 0.f;
#pragma unroll
  for (int j = 0; j < LL; ++j) s += Xi[j];
  float mean = s / flf;
  int off = LL - fl;  // number of leading zeros after push
  float lastX = (fl - 1 >= off) ? (Xi[fl - 1 - off] / mean) : 0.f;
  float lastF = Fi[fl - 1];
  float2 xarr[LL];
#pragma unroll
  for (int j = 0; j < LL; ++j) {
    float px = (j < off) ? 0.f : (Xi[j - off] / mean);
    float pf = (j < off) ? 0.f : Fi[j - off];
    float xf = (px == 0.f) ? lastX : px;
    float ff = (pf == 0.f) ? lastF : pf;
    xarr[j] = make_float2(ff, xf);
  }
  float4* XM = (float4*)(ws + OFF_XM) + (size_t)b * LL;
#pragma unroll
  for (int j = 0; j < LL; ++j) {
    int d = (j == 0) ? 0 : (j - 1);  // m[0]=diffs[0], m[j]=diffs[j-1]
    XM[j] = make_float4(xarr[j].x, xarr[j].y,
                        xarr[d + 1].x - xarr[d].x, xarr[d + 1].y - xarr[d].y);
  }
  float* Z0 = (float*)(ws + OFF_Z0) + (size_t)b * 8;
  const int slot[8] = {0, 1, 4, 5, 2, 3, 6, 7};
#pragma unroll
  for (int q = 0; q < 8; ++q) {
    int h = slot[q];
    Z0[q] = Wi[h * 2] * xarr[0].x + Wi[h * 2 + 1] * xarr[0].y + bi[h];
  }
}

// ---------------- kernel 2: RK4 neural-ODE scan + readout MLP (WMMA) ----------------
__global__ __launch_bounds__(128) void k_ode(const char* __restrict__ ws,
                                             float* __restrict__ out,
                                             const float* __restrict__ br3) {
  const int lane = threadIdx.x & 31;
  const int wid  = threadIdx.x >> 5;
  const int tile = blockIdx.x * 4 + wid;
  const int n    = lane & 15;
  const int hi   = lane >> 4;
  const int row  = tile * 16 + n;

  const float*    W1F  = (const float*)(ws + OFF_W1F);
  const unsigned* B1H  = (const unsigned*)(ws + OFF_B1H);
  const unsigned* W2Fu = (const unsigned*)(ws + OFF_W2F);
  const float*    B2F  = (const float*)(ws + OFF_B2F);
  const float*    WR1F = (const float*)(ws + OFF_WR1F);
  const float*    BR1F = (const float*)(ws + OFF_BR1F);
  const _Float16* WR2F = (const _Float16*)(ws + OFF_WR2F);
  const float*    BR2F = (const float*)(ws + OFF_BR2F);
  const float*    WR3P = (const float*)(ws + OFF_WR3P);
  const float*    Z0   = (const float*)(ws + OFF_Z0);

  // LDS copies of b1 (packed f16) and W2 fragments: keeps VGPR count < 256
  __shared__ unsigned sb1[16 * 32 * 4];   // 8 KB
  __shared__ unsigned sw2[8 * 32 * 8];    // 8 KB (v16h = 8 dwords per (chunk,lane))
  for (int q = threadIdx.x; q < 2048; q += 128) sb1[q] = B1H[q];
  for (int q = threadIdx.x; q < 2048; q += 128) sw2[q] = W2Fu[q];
  __syncthreads();

  // resident fragments: 64 (W1) + 8 (b2) VGPRs
  v2f w1A[32];
#pragma unroll
  for (int q = 0; q < 32; ++q) w1A[q] = *(const v2f*)(W1F + (q * 32 + lane) * 2);
  v8f b2C = *(const v8f*)(B2F + lane * 8);
  v8f Zc = {};

  v4f z = *(const v4f*)(Z0 + (size_t)row * 8 + hi * 4);

  const float4* xm = (const float4*)(ws + OFF_XM) + (size_t)row * LL;

  // Hermite-basis coefficients at fixed s per (substep u, stage st)
  constexpr float C0[2][4] = {{0.f, -1.125f, -1.125f, -1.5f}, {-1.5f, -1.125f, -1.125f, 0.f}};
  constexpr float C1[2][4] = {{1.f, 0.1875f, 0.1875f, -0.25f}, {-0.25f, -0.3125f, -0.3125f, 0.f}};
  constexpr float C3[2][4] = {{0.f, -0.3125f, -0.3125f, -0.25f}, {-0.25f, 0.1875f, 0.1875f, 1.f}};

  auto feval = [&](v4f zin, float dxf, float dxx) -> v4f {
    v2f B0; B0.x = zin.x; B0.y = zin.y;
    v2f B1; B1.x = zin.z; B1.y = zin.w;
    v8f acc = b2C;
#pragma unroll
    for (int c = 0; c < 8; ++c) {
      v8f dE = wmma4(w1A[4 * c + 0], B0, Zc);
      dE = wmma4(w1A[4 * c + 1], B1, dE);
      v8f dO = wmma4(w1A[4 * c + 2], B0, Zc);
      dO = wmma4(w1A[4 * c + 3], B1, dO);
      v4u bE = *(const v4u*)(sb1 + (2 * c * 32 + lane) * 4);
      v4u bO = *(const v4u*)(sb1 + ((2 * c + 1) * 32 + lane) * 4);
      v8u bu;
#pragma unroll
      for (int d = 0; d < 4; ++d) {
        bu[d]     = relu_pack_bias(dE[2 * d], dE[2 * d + 1], bE[d]);
        bu[4 + d] = relu_pack_bias(dO[2 * d], dO[2 * d + 1], bO[d]);
      }
      v16h wf = *(const v16h*)(sw2 + (c * 32 + lane) * 8);
      acc = wmma16(wf, __builtin_bit_cast(v16h, bu), acc);
      if (c & 1) CHUNK_FENCE();   // pin WMMA + DS motion every 2 chunks
    }
    float gg[8];
#pragma unroll
    for (int j = 0; j < 8; ++j) gg[j] = TANHF(acc[j]);
    v4f k;
    k.x = gg[0] * dxf + gg[1] * dxx;
    k.y = gg[2] * dxf + gg[3] * dxx;
    k.z = gg[4] * dxf + gg[5] * dxx;
    k.w = gg[6] * dxf + gg[7] * dxx;
    return k;
  };

  // software-pipelined interval data: one 16B load per iteration, 2 intervals ahead
  float4 cur = xm[0];
  float4 nxt = xm[1];
#pragma unroll 1
  for (int i = 0; i < LL - 1; ++i) {
    int ip = (i < LL - 2) ? (i + 2) : (LL - 1);
    float4 pre = xm[ip];
    float dfx = cur.x - nxt.x, dXx = cur.y - nxt.y;
    float m0f = cur.z, m0x = cur.w, m1f = nxt.z, m1x = nxt.w;
#pragma unroll
    for (int u = 0; u < 2; ++u) {
      v4f acc = {};
      v4f kp  = {};
#pragma unroll
      for (int st = 0; st < 4; ++st) {
        v4f zin = z;
        if (st == 1 || st == 2) zin = z + 0.25f * kp;
        if (st == 3)            zin = z + 0.5f * kp;
        float dxf = C0[u][st] * dfx + C1[u][st] * m0f + C3[u][st] * m1f;
        float dxx = C0[u][st] * dXx + C1[u][st] * m0x + C3[u][st] * m1x;
        v4f k = feval(zin, dxf, dxx);
        if (st == 0)      acc = k;
        else if (st == 3) acc += k;
        else              acc += 2.0f * k;
        kp = k;
      }
      z += (1.f / 12.f) * acc;  // h/6 with h=0.5
    }
    cur = nxt;
    nxt = pre;
  }

  // ---------------- readout: 8 -> 200 (gelu) -> 200 (gelu) -> 1 ----------------
  v2f B0; B0.x = z.x; B0.y = z.y;
  v2f B1; B1.x = z.z; B1.y = z.w;
  auto l1tile = [&](int t) -> v8f {
    v8f r = {};
    if (t > 12) return r;
    v8f d = *(const v8f*)(BR1F + (t * 32 + lane) * 8);
    v2f a0 = *(const v2f*)(WR1F + ((t * 2 + 0) * 32 + lane) * 2);
    v2f a1 = *(const v2f*)(WR1F + ((t * 2 + 1) * 32 + lane) * 2);
    d = wmma4(a0, B0, d);
    d = wmma4(a1, B1, d);
#pragma unroll
    for (int j = 0; j < 8; ++j) r[j] = gelu_exact(d[j]);
    return r;
  };
  v16h bf2[7];
#pragma unroll
  for (int c = 0; c < 7; ++c) {
    v8f e = l1tile(2 * c);
    v8f o = l1tile(2 * c + 1);
    v8u bu;
#pragma unroll
    for (int d = 0; d < 4; ++d) {
      bu[d]     = __builtin_bit_cast(unsigned, __builtin_amdgcn_cvt_pkrtz(e[2 * d], e[2 * d + 1]));
      bu[4 + d] = __builtin_bit_cast(unsigned, __builtin_amdgcn_cvt_pkrtz(o[2 * d], o[2 * d + 1]));
    }
    bf2[c] = __builtin_bit_cast(v16h, bu);
  }
  float partial = 0.f;
#pragma unroll
  for (int t = 0; t < 13; ++t) {
    v8f d2 = *(const v8f*)(BR2F + (t * 32 + lane) * 8);
#pragma unroll
    for (int c = 0; c < 7; ++c) {
      v16h a = *(const v16h*)(WR2F + ((t * 7 + c) * 32 + lane) * 16);
      d2 = wmma16(a, bf2[c], d2);
    }
    CHUNK_FENCE();
    const float* w3 = WR3P + (t * 32 + lane) * 8;
#pragma unroll
    for (int j = 0; j < 8; ++j) partial += gelu_exact(d2[j]) * w3[j];
  }
  partial += __shfl_xor(partial, 16, 32);
  float tv = partial + br3[0];
  if (hi == 0) {
    float T10 = 0.1f + 6.9f / (1.f + expf(-tv));
    out[(size_t)BB * LL + row] = T10;          // T10 output
    out[(size_t)BB * LL + BB + row] = 1.0f;    // M0 output
  }
}

// ---------------- kernel 3: X_out epilogue ----------------
__global__ __launch_bounds__(256) void k_xout(const float* __restrict__ Fa,
                                              const float* __restrict__ TR,
                                              const int* __restrict__ flen,
                                              float* __restrict__ out) {
  int b = blockIdx.x * blockDim.x + threadIdx.x;
  if (b >= BB) return;
  float T10 = out[(size_t)BB * LL + b];
  float E = expf(-TR[b] / T10);
  const float* Fi = Fa + (size_t)b * LL;
  float xo[LL];
  float s = 0.f;
#pragma unroll
  for (int j = 0; j < LL; ++j) {
    float fa = Fi[j];
    float sn, cs;
    sincosf(fa, &sn, &cs);
    float v = (1.f - E) * sn / (1.f - cs * E);
    xo[j] = v;
    s += v;
  }
  float scale = (float)flen[b] / s;
#pragma unroll
  for (int j = 0; j < LL; ++j) out[(size_t)b * LL + j] = xo[j] * scale;
}

// ---------------- host launcher ----------------
extern "C" void kernel_launch(void* const* d_in, const int* in_sizes, int n_in,
                              void* d_out, int out_size, void* d_ws, size_t ws_size,
                              hipStream_t stream) {
  const float* X    = (const float*)d_in[0];
  const float* Fa   = (const float*)d_in[1];
  const float* TR   = (const float*)d_in[2];
  const float* Wi   = (const float*)d_in[3];
  const float* bi   = (const float*)d_in[4];
  const float* W1   = (const float*)d_in[5];
  const float* b1   = (const float*)d_in[6];
  const float* W2   = (const float*)d_in[7];
  const float* b2   = (const float*)d_in[8];
  const float* Wr1  = (const float*)d_in[9];
  const float* br1  = (const float*)d_in[10];
  const float* Wr2  = (const float*)d_in[11];
  const float* br2  = (const float*)d_in[12];
  const float* Wr3  = (const float*)d_in[13];
  const float* br3  = (const float*)d_in[14];
  const int*   flen = (const int*)d_in[16];
  char* ws = (char*)d_ws;
  float* out = (float*)d_out;

  k_prepack<<<32, 256, 0, stream>>>(W1, b1, W2, b2, Wr1, br1, Wr2, br2, Wr3, ws);
  k_prep<<<(BB + 255) / 256, 256, 0, stream>>>(X, Fa, Wi, bi, flen, ws);
  k_ode<<<BB / 64, 128, 0, stream>>>(ws, out, br3);    // 4 waves/block, 16 rows/wave, 256 blocks
  k_xout<<<(BB + 255) / 256, 256, 0, stream>>>(Fa, TR, flen, out);
}
